// FullAttention_17798344474790
// MI455X (gfx1250) — matrix operand order repaired and checked by reading, not compiled, not run on hardware
//
#include <hip/hip_runtime.h>

// ---------------------------------------------------------------------------
// Problem constants (from reference): B=4, C=256, O=256, Cs=32, H=W=64
// ---------------------------------------------------------------------------
#define B_  4
#define C_  256
#define O_  256
#define CS  32
#define H_  64
#define W_  64
#define N_  4096          // H*W
#define KC  2304          // C*9 (im2col K)
#define P_  66            // padded spatial extent
#define LSTRIDE (N_ + 4)  // LDS row stride (pad breaks 64-bank conflict m vs m+8)

typedef __attribute__((ext_vector_type(16))) __bf16 v16bf;
typedef __attribute__((ext_vector_type(8)))  __bf16 v8bf;
typedef __attribute__((ext_vector_type(8)))  float  v8f;
typedef __attribute__((ext_vector_type(4)))  float  v4f;

union ABfrag { v16bf v; v8bf h[2]; };

static __device__ __forceinline__ v8f wmma_bf16(const ABfrag& a, const ABfrag& b, v8f c) {
  // D = A(16x32 bf16) * B(32x16 bf16) + C(16x16 f32)
  return __builtin_amdgcn_wmma_f32_16x16x32_bf16(false, a.v, false, b.v,
                                                 (short)0, c, false, false);
}

// ---------------------------------------------------------------------------
// 1) Pad + convert x (B,C,64,64) f32 -> xpad (B,66,66,C) bf16 channel-last
// ---------------------------------------------------------------------------
__global__ void pad_convert_x(const float* __restrict__ x, __bf16* __restrict__ xp) {
  long idx = (long)blockIdx.x * 256 + threadIdx.x;
  const long total = (long)B_ * P_ * P_ * C_;
  if (idx >= total) return;
  int c  = (int)(idx % C_);
  long t = idx / C_;
  int xx = (int)(t % P_); t /= P_;
  int yy = (int)(t % P_);
  int b  = (int)(t / P_);
  float val = 0.f;
  if (yy >= 1 && yy <= H_ && xx >= 1 && xx <= W_)
    val = x[(((long)b * C_ + c) * H_ + (yy - 1)) * W_ + (xx - 1)];
  xp[idx] = (__bf16)val;
}

// ---------------------------------------------------------------------------
// 2) Weight convert: w (O,C,3,3) f32 -> wt (O, 3*3*C) bf16 with K = kd*256 + c
// ---------------------------------------------------------------------------
__global__ void convert_weights(const float* __restrict__ w, __bf16* __restrict__ wt, int Cout) {
  int idx = blockIdx.x * 256 + threadIdx.x;
  int total = Cout * KC;
  if (idx >= total) return;
  int k  = idx % KC;
  int o  = idx / KC;
  int kd = k / C_;        // dy*3+dx
  int c  = k % C_;
  wt[idx] = (__bf16)w[((long)o * C_ + c) * 9 + kd];
}

// ---------------------------------------------------------------------------
// 3) Implicit-GEMM 3x3 conv via WMMA. One wave computes MT 16x16 tiles
//    (M = out channels, N = 16 consecutive x positions in one row).
//    MODE 0: store bf16 transposed (B,N,Cout) + bias   (q, k)
//    MODE 1: store bf16 planar     (B,Cout,N) + bias   (v)
//    MODE 2: store f32  planar     (B,Cout,N), no bias (res; sigma/bias cancel in BN)
// ---------------------------------------------------------------------------
template<int MT, int MODE>
__global__ void conv_gemm(const __bf16* __restrict__ xp, const __bf16* __restrict__ wt,
                          const float* __restrict__ bias, void* __restrict__ out, int Cout) {
  int wave = blockIdx.x * (blockDim.x >> 5) + (threadIdx.x >> 5);
  int lane = threadIdx.x & 31;
  int Mg   = (Cout >> 4) / MT;                 // M-tile groups
  int nt   = wave & 255;                       // spatial tile (256 per batch)
  int t    = wave >> 8;
  int mg   = t % Mg;
  int b    = t / Mg;
  int y    = nt >> 2;
  int x0   = (nt & 3) << 4;
  int n    = lane & 15;
  int hi   = lane >> 4;                        // half-wave select (K/M split per ISA layout)

  const __bf16* wrow[MT];
  v8f acc[MT];
  v8f zero = {};
  for (int s = 0; s < MT; ++s) {
    wrow[s] = wt + (size_t)((mg * MT + s) * 16 + n) * KC;
    acc[s]  = zero;
  }

  for (int kk = 0; kk < KC / 32; ++kk) {
    int kb = kk << 5;
    int kd = kb >> 8;                          // (dy*3+dx): 256 % 32 == 0 keeps kd uniform
    int cb = kb & 255;
    int dy = kd / 3, dx = kd % 3;
    // B fragment: channels cb+hi*8+0..7 and cb+16+hi*8+0..7 at this spatial tap
    const __bf16* xrow = xp + ((size_t)(b * P_ + (y + dy)) * P_ + (x0 + n + dx)) * C_
                            + cb + hi * 8;
    ABfrag bm;
    bm.h[0] = *(const v8bf*)(xrow);
    bm.h[1] = *(const v8bf*)(xrow + 16);
    __builtin_prefetch((const void*)(xrow + C_), 0, 0);
    for (int s = 0; s < MT; ++s) {
      ABfrag a;
      a.h[0] = *(const v8bf*)(wrow[s] + kb + hi * 8);
      a.h[1] = *(const v8bf*)(wrow[s] + kb + 16 + hi * 8);
      acc[s] = wmma_bf16(a, bm, acc[s]);
    }
  }

  int i = y * W_ + x0 + n;                     // spatial index of this lane's column
  for (int s = 0; s < MT; ++s) {
    int m16 = (mg * MT + s) << 4;
    if (MODE == 0) {                           // (B,N,Cout) bf16, one 16B store per lane
      v8bf o8;
      for (int r = 0; r < 8; ++r)
        o8[r] = (__bf16)(acc[s][r] + bias[m16 + hi * 8 + r]);
      *(v8bf*)((__bf16*)out + ((size_t)b * N_ + i) * Cout + m16 + hi * 8) = o8;
    } else if (MODE == 1) {                    // (B,Cout,N) bf16
      for (int r = 0; r < 8; ++r) {
        int m = m16 + hi * 8 + r;
        ((__bf16*)out)[((size_t)b * Cout + m) * N_ + i] = (__bf16)(acc[s][r] + bias[m]);
      }
    } else {                                   // (B,Cout,N) f32, no bias
      for (int r = 0; r < 8; ++r) {
        int m = m16 + hi * 8 + r;
        ((float*)out)[((size_t)b * Cout + m) * N_ + i] = acc[s][r];
      }
    }
  }
}

// ---------------------------------------------------------------------------
// 4) Fused energy + softmax. One workgroup (8 waves) per (b, i-tile).
//    The 16x4096 f32 score strip lives entirely in LDS (256 KB < 320 KB/WGP):
//      phase 1: each wave computes 32 j-tiles (A fragment loaded ONCE,
//               reused for all 256 WMMAs of the block) -> ds_store to LDS
//      phase 2: row softmax out of LDS, single write of normalized
//               attention to d_out.
//    Saves the full 2x268 MB HBM round trip of a separate softmax pass.
// ---------------------------------------------------------------------------
__global__ void attn_energy_softmax(const __bf16* __restrict__ qt,
                                    const __bf16* __restrict__ kt,
                                    float* __restrict__ att) {
  extern __shared__ float sm[];               // 16 * LSTRIDE floats (dynamic)
  __shared__ float red[256];
  int tid  = threadIdx.x;                     // 0..255
  int wv   = tid >> 5;                        // wave 0..7
  int lane = tid & 31;
  int it = blockIdx.x & 255;
  int b  = blockIdx.x >> 8;
  int n = lane & 15, hi = lane >> 4;
  int i = (it << 4) + n;

  // A fragment: q rows of this i-tile (per 16-bit A 16x32 VGPR layout)
  ABfrag a;
  const __bf16* qrow = qt + ((size_t)b * N_ + i) * CS;
  a.h[0] = *(const v8bf*)(qrow + hi * 8);
  a.h[1] = *(const v8bf*)(qrow + 16 + hi * 8);

  for (int jt = wv; jt < 256; jt += 8) {
    int j = (jt << 4) + n;
    ABfrag bm;
    const __bf16* krow = kt + ((size_t)b * N_ + j) * CS + hi * 8;
    bm.h[0] = *(const v8bf*)(krow);
    bm.h[1] = *(const v8bf*)(krow + 16);
    v8f acc = {};
    acc = wmma_bf16(a, bm, acc);
    // C/D layout: VGPR r -> row (hi*8 + r), column n of this j-tile
    for (int r = 0; r < 8; ++r)
      sm[(hi * 8 + r) * LSTRIDE + (jt << 4) + n] = acc[r];
  }
  __syncthreads();

  // Row softmax over 4096 keys for each of the 16 rows, from LDS
  float* gout = att + ((size_t)b * N_ + (size_t)(it << 4)) * N_;
  for (int r = 0; r < 16; ++r) {
    const float* p = sm + (size_t)r * LSTRIDE;
    float vals[16];
    float mx = -3.402823466e38f;
    for (int u = 0; u < 16; ++u) { vals[u] = p[tid + (u << 8)]; mx = fmaxf(mx, vals[u]); }
    red[tid] = mx; __syncthreads();
    for (int s = 128; s > 0; s >>= 1) {
      if (tid < s) red[tid] = fmaxf(red[tid], red[tid + s]);
      __syncthreads();
    }
    mx = red[0]; __syncthreads();
    float sum = 0.f;
    for (int u = 0; u < 16; ++u) { vals[u] = expf(vals[u] - mx); sum += vals[u]; }
    red[tid] = sum; __syncthreads();
    for (int s = 128; s > 0; s >>= 1) {
      if (tid < s) red[tid] += red[tid + s];
      __syncthreads();
    }
    float inv = 1.0f / red[0]; __syncthreads();
    for (int u = 0; u < 16; ++u)
      gout[(size_t)r * N_ + tid + (u << 8)] = vals[u] * inv;
  }
}

// ---------------------------------------------------------------------------
// 5) BatchNorm statistics per channel (biased var, matches jnp.var)
// ---------------------------------------------------------------------------
__global__ void bn_stats(const float* __restrict__ res, float* __restrict__ stats) {
  __shared__ float s1[256], s2[256];
  int o = blockIdx.x, t = threadIdx.x;
  float sum = 0.f, sq = 0.f;
  for (int b = 0; b < B_; ++b) {
    const float* p = res + ((size_t)b * O_ + o) * N_;
    for (int i = t; i < N_; i += 256) { float v = p[i]; sum += v; sq += v * v; }
  }
  s1[t] = sum; s2[t] = sq; __syncthreads();
  for (int s = 128; s > 0; s >>= 1) {
    if (t < s) { s1[t] += s1[t + s]; s2[t] += s2[t + s]; }
    __syncthreads();
  }
  if (t == 0) {
    const float invn = 1.0f / (float)(B_ * N_);
    float mean = s1[0] * invn;
    float var  = s2[0] * invn - mean * mean;
    stats[o]      = mean;
    stats[O_ + o] = rsqrtf(var + 1e-5f);
  }
}

// ---------------------------------------------------------------------------
// 6) out[b,o,i] = gamma * sum_j v[b,o,j] att[b,i,j] + (1-gamma) * BN(res)[b,o,i]
//    K loop over j (4096) in steps of 32; the 4 waves of a block share the
//    same att rows (L0/L2 reuse) and each wave carries 4 M-tiles reusing the
//    f32->bf16-converted attention fragment.
// ---------------------------------------------------------------------------
__global__ void out_gemm(const __bf16* __restrict__ vmat, const float* __restrict__ att,
                         const float* __restrict__ res, const float* __restrict__ stats,
                         const float* __restrict__ gamma_p, const float* __restrict__ bn_w,
                         const float* __restrict__ bn_b, float* __restrict__ out) {
  int wave = blockIdx.x * (blockDim.x >> 5) + (threadIdx.x >> 5);
  int lane = threadIdx.x & 31;
  int qq = wave & 3;            // which quarter of the 256 output channels
  int t  = wave >> 2;
  int it = t & 255;
  int b  = t >> 8;
  int n = lane & 15, hi = lane >> 4;
  int i = (it << 4) + n;

  const __bf16* vrow[4];
  v8f acc[4];
  v8f zero = {};
  for (int s = 0; s < 4; ++s) {
    vrow[s] = vmat + ((size_t)b * O_ + ((qq * 4 + s) << 4) + n) * N_;
    acc[s]  = zero;
  }
  const float* arow = att + ((size_t)b * N_ + i) * N_ + hi * 8;

  for (int jb = 0; jb < N_; jb += 32) {
    v4f f0 = *(const v4f*)(arow + jb);
    v4f f1 = *(const v4f*)(arow + jb + 4);
    v4f f2 = *(const v4f*)(arow + jb + 16);
    v4f f3 = *(const v4f*)(arow + jb + 20);
    __builtin_prefetch((const void*)(arow + jb + 32), 0, 0);
    ABfrag bm;
    for (int e = 0; e < 4; ++e) {
      bm.h[0][e]     = (__bf16)f0[e];
      bm.h[0][4 + e] = (__bf16)f1[e];
      bm.h[1][e]     = (__bf16)f2[e];
      bm.h[1][4 + e] = (__bf16)f3[e];
    }
    for (int s = 0; s < 4; ++s) {
      ABfrag a;
      a.h[0] = *(const v8bf*)(vrow[s] + jb + hi * 8);
      a.h[1] = *(const v8bf*)(vrow[s] + jb + 16 + hi * 8);
      acc[s] = wmma_bf16(a, bm, acc[s]);
    }
  }

  float g = gamma_p[0];
  for (int s = 0; s < 4; ++s) {
    for (int r = 0; r < 8; ++r) {
      int m = ((qq * 4 + s) << 4) + hi * 8 + r;
      size_t idx = ((size_t)b * O_ + m) * N_ + i;
      float bn = (res[idx] - stats[m]) * stats[O_ + m] * bn_w[m] + bn_b[m];
      out[idx] = g * acc[s][r] + (1.f - g) * bn;
    }
  }
}

// ---------------------------------------------------------------------------
// Host-side orchestration
// ---------------------------------------------------------------------------
extern "C" void kernel_launch(void* const* d_in, const int* in_sizes, int n_in,
                              void* d_out, int out_size, void* d_ws, size_t ws_size,
                              hipStream_t stream) {
  (void)in_sizes; (void)n_in; (void)out_size; (void)ws_size;
  const float* x    = (const float*)d_in[0];
  const float* gam  = (const float*)d_in[1];
  const float* wq   = (const float*)d_in[2];
  const float* bq   = (const float*)d_in[3];
  const float* wk   = (const float*)d_in[4];
  const float* bk   = (const float*)d_in[5];
  const float* wv   = (const float*)d_in[6];
  const float* bv   = (const float*)d_in[7];
  const float* wres = (const float*)d_in[8];
  // d_in[9] (b_res) and d_in[10] (u0): algebraically eliminated by BatchNorm
  const float* bnw  = (const float*)d_in[11];
  const float* bnb  = (const float*)d_in[12];

  float* out = (float*)d_out;
  float* att = out + (size_t)B_ * O_ * N_;      // attention region of d_out

  // carve 256B-aligned scratch from d_ws (~38.4 MB total)
  char* ws = (char*)d_ws;
  auto carve = [&](size_t bytes) {
    char* p = ws;
    ws += (bytes + 255) & ~(size_t)255;
    return p;
  };
  __bf16* xp    = (__bf16*)carve((size_t)B_ * P_ * P_ * C_ * 2);  // 8.9 MB
  __bf16* wq_t  = (__bf16*)carve((size_t)CS * KC * 2);
  __bf16* wk_t  = (__bf16*)carve((size_t)CS * KC * 2);
  __bf16* wv_t  = (__bf16*)carve((size_t)O_ * KC * 2);
  __bf16* wr_t  = (__bf16*)carve((size_t)O_ * KC * 2);
  __bf16* q_t   = (__bf16*)carve((size_t)B_ * N_ * CS * 2);       // 1 MB
  __bf16* k_t   = (__bf16*)carve((size_t)B_ * N_ * CS * 2);       // 1 MB
  __bf16* v_p   = (__bf16*)carve((size_t)B_ * O_ * N_ * 2);       // 8 MB
  float*  res_p = (float*)carve((size_t)B_ * O_ * N_ * 4);        // 16.8 MB
  float*  stats = (float*)carve((size_t)2 * O_ * 4);

  // 1) pad/convert x
  {
    long total = (long)B_ * P_ * P_ * C_;
    pad_convert_x<<<dim3((unsigned)((total + 255) / 256)), dim3(256), 0, stream>>>(x, xp);
  }
  // 2) weights -> bf16 (O,3,3,C)
  convert_weights<<<dim3((CS * KC + 255) / 256), dim3(256), 0, stream>>>(wq, wq_t, CS);
  convert_weights<<<dim3((CS * KC + 255) / 256), dim3(256), 0, stream>>>(wk, wk_t, CS);
  convert_weights<<<dim3((O_ * KC + 255) / 256), dim3(256), 0, stream>>>(wv, wv_t, O_);
  convert_weights<<<dim3((O_ * KC + 255) / 256), dim3(256), 0, stream>>>(wres, wr_t, O_);

  // 3) convs (4 waves / 128-thread block)
  // q,k: Cout=32, MT=2 -> waves = B*1*256 = 1024 -> 256 blocks
  conv_gemm<2, 0><<<dim3(256), dim3(128), 0, stream>>>(xp, wq_t, bq, (void*)q_t, CS);
  conv_gemm<2, 0><<<dim3(256), dim3(128), 0, stream>>>(xp, wk_t, bk, (void*)k_t, CS);
  // v,res: Cout=256, MT=4 -> waves = B*4*256 = 4096 -> 1024 blocks
  conv_gemm<4, 1><<<dim3(1024), dim3(128), 0, stream>>>(xp, wv_t, bv, (void*)v_p, O_);
  conv_gemm<4, 2><<<dim3(1024), dim3(128), 0, stream>>>(xp, wr_t, bv /*unused*/, (void*)res_p, O_);

  // 4) fused energy + softmax: one workgroup per (b, i-tile), 256 KB LDS strip
  {
    size_t shbytes = (size_t)16 * LSTRIDE * sizeof(float);   // 262,400 B < 320 KB
    attn_energy_softmax<<<dim3(B_ * 256), dim3(256), shbytes, stream>>>(q_t, k_t, att);
  }

  // 5) BN stats (one block per channel)
  bn_stats<<<dim3(O_), dim3(256), 0, stream>>>(res_p, stats);

  // 6) PV GEMM + BN + gamma mix: B*256*4 = 4096 waves -> 1024 blocks
  out_gemm<<<dim3(1024), dim3(128), 0, stream>>>(v_p, att, res_p, stats, gam, bnw, bnb, out);
}
